// GNNModel_5463198400699
// MI455X (gfx1250) — matrix operand order repaired and checked by reading, not compiled, not run on hardware
//
#include <hip/hip_runtime.h>

// ---------------- constants ----------------
#define NN     50000
#define EE     800000
#define F_IN   146
#define F_PAD  160
#define DD     256
#define HH     512
#define LL     25
#define EPS_MSG 1e-7f
#define TT      0.01f

typedef __attribute__((ext_vector_type(16))) __bf16 v16bf;
typedef __attribute__((ext_vector_type(8)))  float  v8f;
typedef __attribute__((ext_vector_type(4)))  int    v4i;

union F16x16 { uint4 u[2]; v16bf v; };

#define AS1 __attribute__((address_space(1)))
#define AS3 __attribute__((address_space(3)))

#if defined(__has_builtin)
#if __has_builtin(__builtin_amdgcn_global_load_async_to_lds_b128) && \
    __has_builtin(__builtin_amdgcn_s_wait_asynccnt)
#define USE_ASYNC_LDS 1
#endif
#endif
#ifndef USE_ASYNC_LDS
#define USE_ASYNC_LDS 0
#endif

// float -> bf16 (round to nearest even), stored as ushort
__device__ __forceinline__ unsigned short f2bf(float f) {
    unsigned u = __float_as_uint(f);
    u += 0x7FFFu + ((u >> 16) & 1u);
    return (unsigned short)(u >> 16);
}

// ---------------- weight prep ----------------
// Transposed bf16 conversion: out[l][n][k] = in[l][k][n]   (weights stored N-major)
__global__ void f2bf_T_kernel(const float* __restrict__ in, unsigned short* __restrict__ out,
                              int K, int Nc, int layers) {
    int i = blockIdx.x * blockDim.x + threadIdx.x;
    int per = K * Nc;
    if (i >= per * layers) return;
    int l = i / per, r = i - l * per;
    int n = r / K, k = r - n * K;
    out[i] = f2bf(in[(size_t)l * per + (size_t)k * Nc + n]);
}

// x: N x 146 -> bf16 N x 160, with cols >=121 scaled x/100-0.5, cols >=146 zero
__global__ void prep_x_kernel(const float* __restrict__ x, unsigned short* __restrict__ xp) {
    int i = blockIdx.x * blockDim.x + threadIdx.x;
    if (i >= NN * F_PAD) return;
    int n = i / F_PAD, f = i - n * F_PAD;
    float v = 0.f;
    if (f < F_IN) {
        v = x[(size_t)n * F_IN + f];
        if (f >= 121) v = v * 0.01f - 0.5f;
    }
    xp[i] = f2bf(v);
}

// enc_W: 146 x 256 -> transposed bf16 256 x 160 zero padded: out[n][f]
__global__ void prep_encW_kernel(const float* __restrict__ w, unsigned short* __restrict__ wp) {
    int i = blockIdx.x * blockDim.x + threadIdx.x;
    if (i >= DD * F_PAD) return;
    int n = i / F_PAD, f = i - n * F_PAD;
    wp[i] = (f < F_IN) ? f2bf(w[(size_t)f * DD + n]) : (unsigned short)0;
}

// ---------------- WMMA bf16 GEMM ----------------
// C(MxNc) = [res +] A(MxK,row-major bf16) @ Bt(NcxK, N-major bf16)^T + bias, optional relu
// 128x64 C tile per 256-thread block (8 waves; wave w: rows w*16..+15, all 64 cols).
__global__ __launch_bounds__(256)
void gemm_bf16_kernel(const unsigned short* __restrict__ A, int lda,
                      const unsigned short* __restrict__ Bt,
                      const float* __restrict__ bias,
                      const float* residual,   // nullable; may alias C
                      float* C,
                      int M, int Nc, int K, int relu)
{
    const int tid  = threadIdx.x;
    const int wave = tid >> 5;
    const int lane = tid & 31;
    const int half = lane >> 4;
    const int l16  = lane & 15;
    const int tileM = blockIdx.y * 128;
    const int tileN = blockIdx.x * 64;

    v8f acc[4] = {};

    // per-thread tile slots (row stride 40 ushorts = 80 B: 16B-aligned + conflict padding)
    const int ar0 = (tid * 2)     >> 2, as0 = (tid * 2)     & 3;
    const int ar1 = (tid * 2 + 1) >> 2, as1 = (tid * 2 + 1) & 3;
    const int br  = tid >> 2,           bs  = tid & 3;
    const int ksteps = K >> 5;

#if USE_ASYNC_LDS
    // ---------- async global->LDS double-buffered pipeline (ASYNCcnt path) ----------
    __shared__ __align__(16) unsigned short As[2][128 * 40];
    __shared__ __align__(16) unsigned short Bs[2][64 * 40];

    // zero-fill M-tail rows once (async copies are EXEC-masked off for them)
    {
        const uint4 zz = make_uint4(0, 0, 0, 0);
        if (tileM + ar0 >= M) {
            *(uint4*)(&As[0][ar0 * 40 + as0 * 8]) = zz;
            *(uint4*)(&As[1][ar0 * 40 + as0 * 8]) = zz;
        }
        if (tileM + ar1 >= M) {
            *(uint4*)(&As[1][ar1 * 40 + as1 * 8]) = zz;
            *(uint4*)(&As[0][ar1 * 40 + as1 * 8]) = zz;
        }
    }

    auto issue = [&](int buf, int k0) {
        int g0 = tileM + ar0, g1 = tileM + ar1;
        if (g0 < M)
            __builtin_amdgcn_global_load_async_to_lds_b128(
                (AS1 v4i*)(A + (size_t)g0 * lda + k0 + as0 * 8),
                (AS3 v4i*)(&As[buf][ar0 * 40 + as0 * 8]), 0, 0);
        if (g1 < M)
            __builtin_amdgcn_global_load_async_to_lds_b128(
                (AS1 v4i*)(A + (size_t)g1 * lda + k0 + as1 * 8),
                (AS3 v4i*)(&As[buf][ar1 * 40 + as1 * 8]), 0, 0);
        __builtin_amdgcn_global_load_async_to_lds_b128(
            (AS1 v4i*)(Bt + (size_t)(tileN + br) * K + k0 + bs * 8),
            (AS3 v4i*)(&Bs[buf][br * 40 + bs * 8]), 0, 0);
    };

    issue(0, 0);
    __builtin_amdgcn_s_wait_asynccnt(0);
    __syncthreads();

    for (int kt = 0; kt < ksteps; ++kt) {
        const int cur = kt & 1;
        const bool more = (kt + 1) < ksteps;
        if (more) issue(cur ^ 1, (kt + 1) * 32);   // overlap async DMA with WMMA

        F16x16 af;
        {
            const unsigned short* ap = &As[cur][(wave * 16 + l16) * 40];
            af.u[0] = *(const uint4*)(ap + half * 8);
            af.u[1] = *(const uint4*)(ap + 16 + half * 8);
        }
#pragma unroll
        for (int c = 0; c < 4; ++c) {
            F16x16 bfr;
            const unsigned short* bp = &Bs[cur][(c * 16 + l16) * 40 + half * 8];
            bfr.u[0] = *(const uint4*)(bp);
            bfr.u[1] = *(const uint4*)(bp + 16);
            acc[c] = __builtin_amdgcn_wmma_f32_16x16x32_bf16(
                         false, af.v, false, bfr.v, (short)0, acc[c], false, false);
        }
        if (more) {
            __builtin_amdgcn_s_wait_asynccnt(0);
            __syncthreads();
        }
    }
#else
    // ---------- fallback: register-staged pipeline ----------
    __shared__ __align__(16) unsigned short As[128 * 40];
    __shared__ __align__(16) unsigned short Bs[64 * 40];
    uint4 ra0, ra1, rb;

    auto loadG = [&](int k0) {
        int g0 = tileM + ar0, g1 = tileM + ar1;
        ra0 = (g0 < M) ? *(const uint4*)(A + (size_t)g0 * lda + k0 + as0 * 8)
                       : make_uint4(0, 0, 0, 0);
        ra1 = (g1 < M) ? *(const uint4*)(A + (size_t)g1 * lda + k0 + as1 * 8)
                       : make_uint4(0, 0, 0, 0);
        rb  = *(const uint4*)(Bt + (size_t)(tileN + br) * K + k0 + bs * 8);
    };
    auto storeLDS = [&]() {
        *(uint4*)(&As[ar0 * 40 + as0 * 8]) = ra0;
        *(uint4*)(&As[ar1 * 40 + as1 * 8]) = ra1;
        *(uint4*)(&Bs[br * 40 + bs * 8])   = rb;
    };

    loadG(0);
    storeLDS();
    __syncthreads();

    for (int kt = 0; kt < ksteps; ++kt) {
        const bool more = (kt + 1) < ksteps;
        if (more) loadG((kt + 1) * 32);
        if (kt + 2 < ksteps) {
            __builtin_prefetch(A + (size_t)(tileM + ar0) * lda + (kt + 2) * 32, 0, 1);
            __builtin_prefetch(Bt + (size_t)(tileN + br) * K + (kt + 2) * 32, 0, 1);
        }
        F16x16 af;
        {
            const unsigned short* ap = &As[(wave * 16 + l16) * 40];
            af.u[0] = *(const uint4*)(ap + half * 8);
            af.u[1] = *(const uint4*)(ap + 16 + half * 8);
        }
#pragma unroll
        for (int c = 0; c < 4; ++c) {
            F16x16 bfr;
            const unsigned short* bp = &Bs[(c * 16 + l16) * 40 + half * 8];
            bfr.u[0] = *(const uint4*)(bp);
            bfr.u[1] = *(const uint4*)(bp + 16);
            acc[c] = __builtin_amdgcn_wmma_f32_16x16x32_bf16(
                         false, af.v, false, bfr.v, (short)0, acc[c], false, false);
        }
        __syncthreads();
        if (more) { storeLDS(); __syncthreads(); }
    }
#endif

    // C/D layout: VGPR v -> M = v + 8*half ; N = l16
#pragma unroll
    for (int c = 0; c < 4; ++c) {
        const int col = tileN + c * 16 + l16;
#pragma unroll
        for (int v = 0; v < 8; ++v) {
            int row = tileM + wave * 16 + v + half * 8;
            if (row < M) {
                float val = acc[c][v] + bias[col];
                if (residual) val += residual[(size_t)row * Nc + col];
                if (relu) val = fmaxf(val, 0.f);
                C[(size_t)row * Nc + col] = val;
            }
        }
    }
}

// ---------------- pre-norm: z = relu(LN(h)) (or z = h), zero agg buffers ----------------
__global__ __launch_bounds__(256)
void prenorm_kernel(const float* __restrict__ h,
                    const float* __restrict__ g, const float* __restrict__ b,
                    float* __restrict__ z, unsigned short* zb,
                    float* m0, float* d0, float* w0, int applyLN)
{
    const int n = blockIdx.x, t = threadIdx.x;
    const size_t idx = (size_t)n * DD + t;
    float v = h[idx], out;
    if (applyLN) {
        __shared__ float red[256];
        red[t] = v; __syncthreads();
        for (int s = 128; s > 0; s >>= 1) { if (t < s) red[t] += red[t + s]; __syncthreads(); }
        float mu = red[0] * (1.f / DD);
        __syncthreads();
        float dv = v - mu;
        red[t] = dv * dv; __syncthreads();
        for (int s = 128; s > 0; s >>= 1) { if (t < s) red[t] += red[t + s]; __syncthreads(); }
        float inv = rsqrtf(red[0] * (1.f / DD) + 1e-5f);
        out = fmaxf(dv * inv * g[t] + b[t], 0.f);
    } else {
        out = v;
    }
    z[idx] = out;
    if (zb) zb[idx] = f2bf(out);
    if (m0) { m0[idx] = 0.f; d0[idx] = 0.f; w0[idx] = 0.f; }
}

// ---------------- LN(512) + relu -> bf16 ----------------
__global__ __launch_bounds__(256)
void lnrelu512_kernel(const float* __restrict__ hp,
                      const float* __restrict__ g, const float* __restrict__ b,
                      unsigned short* __restrict__ ob)
{
    const int n = blockIdx.x, t = threadIdx.x;
    const size_t base = (size_t)n * HH;
    float v0 = hp[base + t], v1 = hp[base + t + 256];
    __shared__ float red[256];
    red[t] = v0 + v1; __syncthreads();
    for (int s = 128; s > 0; s >>= 1) { if (t < s) red[t] += red[t + s]; __syncthreads(); }
    float mu = red[0] * (1.f / HH);
    __syncthreads();
    float a0 = v0 - mu, a1 = v1 - mu;
    red[t] = a0 * a0 + a1 * a1; __syncthreads();
    for (int s = 128; s > 0; s >>= 1) { if (t < s) red[t] += red[t + s]; __syncthreads(); }
    float inv = rsqrtf(red[0] * (1.f / HH) + 1e-5f);
    ob[base + t]       = f2bf(fmaxf(a0 * inv * g[t] + b[t], 0.f));
    ob[base + t + 256] = f2bf(fmaxf(a1 * inv * g[t + 256] + b[t + 256], 0.f));
}

// ---------------- edge pass 1: segment max of logit (positive -> uint atomicMax) --------
__global__ __launch_bounds__(256)
void edge_max_kernel(const int* __restrict__ src, const int* __restrict__ dst,
                     const float* __restrict__ z, unsigned int* __restrict__ m)
{
    int idx = blockIdx.x * blockDim.x + threadIdx.x;
    if (idx >= EE * 64) return;
    int e = idx >> 6, c = (idx & 63) * 4;
    int s = src[e], d = dst[e];
    float4 zv = *(const float4*)(z + (size_t)s * DD + c);
    unsigned int* mp = m + (size_t)d * DD + c;
    atomicMax(mp + 0, __float_as_uint((fmaxf(zv.x, 0.f) + EPS_MSG) * TT));
    atomicMax(mp + 1, __float_as_uint((fmaxf(zv.y, 0.f) + EPS_MSG) * TT));
    atomicMax(mp + 2, __float_as_uint((fmaxf(zv.z, 0.f) + EPS_MSG) * TT));
    atomicMax(mp + 3, __float_as_uint((fmaxf(zv.w, 0.f) + EPS_MSG) * TT));
}

// ---------------- edge pass 2: den += num ; wsum += num*msg ----------------
__global__ __launch_bounds__(256)
void edge_sum_kernel(const int* __restrict__ src, const int* __restrict__ dst,
                     const float* __restrict__ z, const unsigned int* __restrict__ m,
                     float* __restrict__ den, float* __restrict__ wsum)
{
    int idx = blockIdx.x * blockDim.x + threadIdx.x;
    if (idx >= EE * 64) return;
    int e = idx >> 6, c = (idx & 63) * 4;
    int s = src[e], d = dst[e];
    float4 zv = *(const float4*)(z + (size_t)s * DD + c);
    const size_t o = (size_t)d * DD + c;
#pragma unroll
    for (int j = 0; j < 4; ++j) {
        float zj = (j == 0) ? zv.x : (j == 1) ? zv.y : (j == 2) ? zv.z : zv.w;
        float msg = fmaxf(zj, 0.f) + EPS_MSG;
        float num = __expf(msg * TT - __uint_as_float(m[o + j]));
        atomicAdd(den + o + j, num);
        atomicAdd(wsum + o + j, num * msg);
    }
}

// ---------------- node finish: hin = wsum/(den+1e-16) + z -> bf16 ----------------
__global__ __launch_bounds__(256)
void node_finish_kernel(const float* __restrict__ wsum, const float* __restrict__ den,
                        const float* __restrict__ z, unsigned short* __restrict__ hinb)
{
    int i = blockIdx.x * blockDim.x + threadIdx.x;
    if (i >= NN * DD) return;
    hinb[i] = f2bf(wsum[i] / (den[i] + 1e-16f) + z[i]);
}

// ---------------- head: out = fc1o @ out_W + out_b ----------------
__global__ __launch_bounds__(256)
void out_kernel(const float* __restrict__ fo, const float* __restrict__ ow,
                const float* __restrict__ obias, float* __restrict__ out)
{
    int n = blockIdx.x * blockDim.x + threadIdx.x;
    if (n >= NN) return;
    float a = 0.f;
#pragma unroll 8
    for (int k = 0; k < 128; ++k) a += fo[(size_t)n * 128 + k] * ow[k];
    out[n] = a + obias[0];
}

// ================= host launch =================
extern "C" void kernel_launch(void* const* d_in, const int* in_sizes, int n_in,
                              void* d_out, int out_size, void* d_ws, size_t ws_size,
                              hipStream_t stream)
{
    const float* x       = (const float*)d_in[0];
    const int*   eidx    = (const int*)d_in[1];
    const float* enc_W   = (const float*)d_in[2];
    const float* enc_b   = (const float*)d_in[3];
    const float* conv_W1 = (const float*)d_in[4];
    const float* conv_b1 = (const float*)d_in[5];
    const float* cln_g   = (const float*)d_in[6];
    const float* cln_b   = (const float*)d_in[7];
    const float* conv_W2 = (const float*)d_in[8];
    const float* conv_b2 = (const float*)d_in[9];
    const float* rln_g   = (const float*)d_in[10];
    const float* rln_b   = (const float*)d_in[11];
    const float* fc1_W   = (const float*)d_in[12];
    const float* fc1_b   = (const float*)d_in[13];
    const float* out_W   = (const float*)d_in[14];
    const float* out_b   = (const float*)d_in[15];
    float* out = (float*)d_out;

    const int* src = eidx;
    const int* dst = eidx + EE;

    // workspace carve-out
    char* p = (char*)d_ws;
    auto alloc = [&](size_t bytes) -> char* {
        char* r = p; p += (bytes + 255) & ~(size_t)255; return r;
    };
    float* h      = (float*)alloc((size_t)NN * DD * 4);
    float* z      = (float*)alloc((size_t)NN * DD * 4);
    float* mbuf   = (float*)alloc((size_t)NN * DD * 4);
    float* den    = (float*)alloc((size_t)NN * DD * 4);
    float* wsum   = (float*)alloc((size_t)NN * DD * 4);
    float* hidden = (float*)alloc((size_t)NN * HH * 4);
    float* fc1o   = (float*)alloc((size_t)NN * 128 * 4);
    unsigned short* hinb  = (unsigned short*)alloc((size_t)NN * DD * 2);
    unsigned short* hidb  = (unsigned short*)alloc((size_t)NN * HH * 2);
    unsigned short* xpad  = (unsigned short*)alloc((size_t)NN * F_PAD * 2);
    unsigned short* encWp = (unsigned short*)alloc((size_t)DD * F_PAD * 2);   // 256 x 160 (transposed)
    unsigned short* W1b   = (unsigned short*)alloc((size_t)LL * DD * HH * 2); // [l][512][256]
    unsigned short* W2b   = (unsigned short*)alloc((size_t)LL * HH * DD * 2); // [l][256][512]
    unsigned short* fWb   = (unsigned short*)alloc((size_t)DD * 128 * 2);     // [128][256]

    const int MT = (NN + 127) / 128;
    const dim3 egrid((EE * 64 + 255) / 256);

    // ---- weight prep (bf16, transposed to N-major) ----
    prep_x_kernel   <<<(NN * F_PAD + 255) / 256, 256, 0, stream>>>(x, xpad);
    prep_encW_kernel<<<(DD * F_PAD + 255) / 256, 256, 0, stream>>>(enc_W, encWp);
    f2bf_T_kernel<<<(LL * DD * HH + 255) / 256, 256, 0, stream>>>(conv_W1, W1b, DD, HH, LL);
    f2bf_T_kernel<<<(LL * HH * DD + 255) / 256, 256, 0, stream>>>(conv_W2, W2b, HH, DD, LL);
    f2bf_T_kernel<<<(DD * 128 + 255) / 256, 256, 0, stream>>>(fc1_W, fWb, DD, 128, 1);

    // ---- encoder: h = xpad @ encWp^T + enc_b  (K padded to 160) ----
    gemm_bf16_kernel<<<dim3(DD / 64, MT), 256, 0, stream>>>(
        xpad, F_PAD, encWp, enc_b, nullptr, h, NN, DD, F_PAD, 0);

    // ---- 25 GENConv layers ----
    for (int l = 0; l < LL; ++l) {
        // z = (l==0) ? h : relu(LN(h, res_ln[l])); zero m/den/wsum
        prenorm_kernel<<<NN, 256, 0, stream>>>(
            h, rln_g + (size_t)l * DD, rln_b + (size_t)l * DD,
            z, nullptr, mbuf, den, wsum, l > 0 ? 1 : 0);

        edge_max_kernel<<<egrid, 256, 0, stream>>>(src, dst, z, (unsigned int*)mbuf);
        edge_sum_kernel<<<egrid, 256, 0, stream>>>(src, dst, z, (const unsigned int*)mbuf, den, wsum);
        node_finish_kernel<<<(NN * DD + 255) / 256, 256, 0, stream>>>(wsum, den, z, hinb);

        // hidden = hin @ W1[l] + b1[l]
        gemm_bf16_kernel<<<dim3(HH / 64, MT), 256, 0, stream>>>(
            hinb, DD, W1b + (size_t)l * DD * HH,
            conv_b1 + (size_t)l * HH, nullptr, hidden, NN, HH, DD, 0);
        // hidb = bf16(relu(LN(hidden, conv_ln[l])))
        lnrelu512_kernel<<<NN, 256, 0, stream>>>(
            hidden, cln_g + (size_t)l * HH, cln_b + (size_t)l * HH, hidb);
        // h = (l>0 ? h : 0) + hidb @ W2[l] + b2[l]
        gemm_bf16_kernel<<<dim3(DD / 64, MT), 256, 0, stream>>>(
            hidb, HH, W2b + (size_t)l * HH * DD,
            conv_b2 + (size_t)l * DD, (l > 0) ? h : nullptr, h, NN, DD, HH, 0);
    }

    // ---- head ----
    prenorm_kernel<<<NN, 256, 0, stream>>>(h, rln_g, rln_b, z, hinb,
                                           nullptr, nullptr, nullptr, 1);
    gemm_bf16_kernel<<<dim3(128 / 64, MT), 256, 0, stream>>>(
        hinb, DD, fWb, fc1_b, nullptr, fc1o, NN, 128, DD, 1);
    out_kernel<<<(NN + 255) / 256, 256, 0, stream>>>(fc1o, out_W, out_b, out);
}
